// PINN_50766513438747
// MI455X (gfx1250) — compile-verified, matrix operand order checked by reading, not run on hardware
//
#include <hip/hip_runtime.h>

typedef __attribute__((ext_vector_type(16))) _Float16 v16h;
typedef __attribute__((ext_vector_type(8)))  float    v8f;

#define PINN_NU 0.1f
#define RES_OFF 32768   // float offset of per-point result buffer inside d_ws (128KB)

// gfx1250 has a hardware tanh (TRANS op, co-executes with WMMA); use it if the
// toolchain exposes the builtin, else fall back to libm.
#if defined(__AMDGCN__) && __has_builtin(__builtin_amdgcn_tanhf)
__device__ __forceinline__ float fast_tanh(float x) { return __builtin_amdgcn_tanhf(x); }
#else
__device__ __forceinline__ float fast_tanh(float x) { return tanhf(x); }
#endif

// ---------------------------------------------------------------------------
// D = A(16x32 f16 weights) * B(32x16 f16 activations) + C (f32)
// ---------------------------------------------------------------------------
__device__ __forceinline__ v8f wmma_f16(v16h a, v16h b, v8f c) {
  return __builtin_amdgcn_wmma_f32_16x16x32_f16(false, a, false, b,
                                                (short)0, c, false, false);
}

// One 16-row output tile r of a 128x128 layer: 4 WMMAs along K.
__device__ __forceinline__ v8f mm_tile(const v16h* __restrict__ wl, int lane,
                                       const v16h q[4], int r) {
  v8f acc = {0.f, 0.f, 0.f, 0.f, 0.f, 0.f, 0.f, 0.f};
#pragma unroll
  for (int k = 0; k < 4; ++k)
    acc = wmma_f16(wl[(r * 4 + k) * 32 + lane], q[k], acc);
  return acc;
}

__device__ __forceinline__ float xsum16(float v) { return v + __shfl_xor(v, 16, 32); }

// ---------------------------------------------------------------------------
// prep: swizzle W1..W3 (128x128 f32, row-major [out][in]) into the WMMA
// A-operand f16 layout.
// ---------------------------------------------------------------------------
__global__ void pinn_prep_weights(const float* __restrict__ W1,
                                  const float* __restrict__ W2,
                                  const float* __restrict__ W3,
                                  _Float16* __restrict__ ws) {
  int idx = blockIdx.x * blockDim.x + threadIdx.x;   // 49152 total
  int l = idx >> 14, rem = idx & 16383;
  int tile = rem >> 9, rem2 = rem & 511;
  int lane = rem2 >> 4, e = rem2 & 15;
  int r = tile >> 2, k = tile & 3;
  int row = r * 16 + (lane & 15);
  int col = k * 32 + 16 * (e >> 3) + (e & 7) + 8 * ((lane >> 4) & 1);
  const float* W = (l == 0) ? W1 : (l == 1) ? W2 : W3;
  ws[idx] = (_Float16)W[row * 128 + col];
}

__global__ void pinn_zero_out(float* out) {
  if (threadIdx.x == 0 && blockIdx.x == 0) out[0] = 0.f;
}

// ---------------------------------------------------------------------------
// Shared LDS staging used by the heavy kernels.
// ---------------------------------------------------------------------------
struct PinnLds {
  alignas(32) _Float16 w_swz[49152];   // 3 layers * 32KB swizzled f16
  float w0_l[384], b0_l[128], bh_l[384], w4_l[384], b4_l[3], red[8];
};

__device__ __forceinline__ void pinn_stage(PinnLds& S, int tid,
                                           const _Float16* __restrict__ wsw,
                                           const float* __restrict__ W0,
                                           const float* __restrict__ b0,
                                           const float* __restrict__ b1,
                                           const float* __restrict__ b2,
                                           const float* __restrict__ b3,
                                           const float* __restrict__ W4,
                                           const float* __restrict__ b4) {
  const uint4* s = (const uint4*)wsw;
  uint4* d = (uint4*)S.w_swz;
  for (int i = tid; i < 6144; i += 256) d[i] = s[i];
  for (int i = tid; i < 384; i += 256) {
    S.w0_l[i] = W0[i];
    S.w4_l[i] = W4[i];
    S.bh_l[i] = (i < 128) ? b1[i] : (i < 256) ? b2[i - 128] : b3[i - 256];
  }
  for (int i = tid; i < 128; i += 256) S.b0_l[i] = b0[i];
  if (tid < 3) S.b4_l[tid] = b4[tid];
  __syncthreads();
}

// value pass for one hidden layer: qv <- tanh(W*qv + b)   (32 WMMAs)
__device__ __forceinline__ void value_layer(const PinnLds& S, const v16h* wl,
                                            int lane, int h, int L, v16h qv[4]) {
  v16h nv[4];
#pragma unroll
  for (int r = 0; r < 8; ++r) {
    v8f z = mm_tile(wl, lane, qv, r);
#pragma unroll
    for (int v = 0; v < 8; ++v) {
      const int row = 16 * r + v + 8 * h;
      nv[r >> 1][((r & 1) << 3) + v] =
          (_Float16)fast_tanh(z[v] + S.bh_l[L * 128 + row]);
    }
  }
#pragma unroll
  for (int c = 0; c < 4; ++c) qv[c] = nv[c];
}

// ---------------------------------------------------------------------------
// PDE pass for one spatial tangent kk (1 or 2): propagate value + J_kk + H_kk,
// store 5 per-point dot products into res[p*16 + (kk-1)*5 ...].
// 96 persistent VGPRs (qv,qj,qh) -> no spills.
// ---------------------------------------------------------------------------
__launch_bounds__(256)
__global__ void pinn_jh_kernel(const float* __restrict__ xp,
                               const float* __restrict__ W0, const float* __restrict__ b0,
                               const float* __restrict__ b1, const float* __restrict__ b2,
                               const float* __restrict__ b3,
                               const float* __restrict__ W4, const float* __restrict__ b4,
                               const _Float16* __restrict__ wsw,
                               float* __restrict__ res, int kk) {
  __shared__ PinnLds S;
  const int tid = threadIdx.x;
  pinn_stage(S, tid, wsw, W0, b0, b1, b2, b3, W4, b4);

  const int lane = tid & 31, wid = tid >> 5;
  const int h = lane >> 4, pcol = lane & 15;
  const int gwave = blockIdx.x * 8 + wid;
  const int nw = gridDim.x * 8;
  const int base = (kk - 1) * 5;

  for (int g = gwave; g < 4096; g += nw) {    // 4096 groups / 512 waves = 8 each
    const int p = g * 16 + pcol;
    __builtin_prefetch(&xp[(p + nw * 16) * 3], 0, 1);
    float x0 = xp[p * 3 + 0], x1 = xp[p * 3 + 1], x2 = xp[p * 3 + 2];

    v16h qv[4], qj[4], qh[4];
#pragma unroll
    for (int c = 0; c < 4; ++c)
#pragma unroll
      for (int e = 0; e < 16; ++e) {
        const int row = 32 * c + 16 * (e >> 3) + (e & 7) + 8 * h;
        float wk = S.w0_l[row * 3 + kk];
        float z = S.w0_l[row * 3 + 0] * x0 + S.w0_l[row * 3 + 1] * x1
                + S.w0_l[row * 3 + 2] * x2 + S.b0_l[row];
        float a = fast_tanh(z), s = 1.f - a * a;
        qv[c][e] = (_Float16)a;
        qj[c][e] = (_Float16)(s * wk);
        qh[c][e] = (_Float16)(-2.f * a * s * wk * wk);
      }

    for (int L = 0; L < 3; ++L) {
      const v16h* wl = (const v16h*)&S.w_swz[L * 16384];
      value_layer(S, wl, lane, h, L, qv);
      v16h nj[4], nh[4];
#pragma unroll
      for (int r = 0; r < 8; ++r) {
        v8f zj = mm_tile(wl, lane, qj, r);
        v8f zh = mm_tile(wl, lane, qh, r);
#pragma unroll
        for (int v = 0; v < 8; ++v) {
          const int e = ((r & 1) << 3) + v;
          float a = (float)qv[r >> 1][e];
          float s = 1.f - a * a;
          float jz = zj[v], hz = zh[v];
          nj[r >> 1][e] = (_Float16)(s * jz);
          nh[r >> 1][e] = (_Float16)(s * hz - 2.f * a * s * jz * jz);
        }
      }
#pragma unroll
      for (int c = 0; c < 4; ++c) { qj[c] = nj[c]; qh[c] = nh[c]; }
    }

    // dots: d_{o,kk} = W4[o].J (o=0,1,2), g_{o,kk} = W4[o].H (o=0,1)
    float dj0 = 0, dj1 = 0, dj2 = 0, gh0 = 0, gh1 = 0;
#pragma unroll
    for (int c = 0; c < 4; ++c)
#pragma unroll
      for (int e = 0; e < 16; ++e) {
        const int row = 32 * c + 16 * (e >> 3) + (e & 7) + 8 * h;
        float w40 = S.w4_l[row], w41 = S.w4_l[128 + row], w42 = S.w4_l[256 + row];
        float jv = (float)qj[c][e], hv = (float)qh[c][e];
        dj0 += w40 * jv; dj1 += w41 * jv; dj2 += w42 * jv;
        gh0 += w40 * hv; gh1 += w41 * hv;
      }
    dj0 = xsum16(dj0); dj1 = xsum16(dj1); dj2 = xsum16(dj2);
    gh0 = xsum16(gh0); gh1 = xsum16(gh1);
    if (h == 0) {
      res[p * 16 + base + 0] = dj0;
      res[p * 16 + base + 1] = dj1;
      res[p * 16 + base + 2] = dj2;
      res[p * 16 + base + 3] = gh0;
      res[p * 16 + base + 4] = gh1;
    }
  }
}

// ---------------------------------------------------------------------------
// Final PDE pass: value + time tangent J0; combine with stored results to
// produce momentum + divergence losses.
// ---------------------------------------------------------------------------
__launch_bounds__(256)
__global__ void pinn_j0_kernel(const float* __restrict__ xp,
                               const float* __restrict__ W0, const float* __restrict__ b0,
                               const float* __restrict__ b1, const float* __restrict__ b2,
                               const float* __restrict__ b3,
                               const float* __restrict__ W4, const float* __restrict__ b4,
                               const _Float16* __restrict__ wsw,
                               const float* __restrict__ res, float* out) {
  __shared__ PinnLds S;
  const int tid = threadIdx.x;
  pinn_stage(S, tid, wsw, W0, b0, b1, b2, b3, W4, b4);

  const int lane = tid & 31, wid = tid >> 5;
  const int h = lane >> 4, pcol = lane & 15;
  const int gwave = blockIdx.x * 8 + wid;
  const int nw = gridDim.x * 8;
  float tsum = 0.f;

  for (int g = gwave; g < 4096; g += nw) {
    const int p = g * 16 + pcol;
    __builtin_prefetch(&xp[(p + nw * 16) * 3], 0, 1);
    float x0 = xp[p * 3 + 0], x1 = xp[p * 3 + 1], x2 = xp[p * 3 + 2];

    v16h qv[4], qj[4];
#pragma unroll
    for (int c = 0; c < 4; ++c)
#pragma unroll
      for (int e = 0; e < 16; ++e) {
        const int row = 32 * c + 16 * (e >> 3) + (e & 7) + 8 * h;
        float wk = S.w0_l[row * 3 + 0];
        float z = S.w0_l[row * 3 + 0] * x0 + S.w0_l[row * 3 + 1] * x1
                + S.w0_l[row * 3 + 2] * x2 + S.b0_l[row];
        float a = fast_tanh(z), s = 1.f - a * a;
        qv[c][e] = (_Float16)a;
        qj[c][e] = (_Float16)(s * wk);
      }

    for (int L = 0; L < 3; ++L) {
      const v16h* wl = (const v16h*)&S.w_swz[L * 16384];
      value_layer(S, wl, lane, h, L, qv);
      v16h nj[4];
#pragma unroll
      for (int r = 0; r < 8; ++r) {
        v8f zj = mm_tile(wl, lane, qj, r);
#pragma unroll
        for (int v = 0; v < 8; ++v) {
          const int e = ((r & 1) << 3) + v;
          float a = (float)qv[r >> 1][e];
          nj[r >> 1][e] = (_Float16)((1.f - a * a) * zj[v]);
        }
      }
#pragma unroll
      for (int c = 0; c < 4; ++c) qj[c] = nj[c];
    }

    // dots: y0,y1 (value) and d00,d10 (time tangent)
    float y0 = 0, y1 = 0, d00 = 0, d10 = 0;
#pragma unroll
    for (int c = 0; c < 4; ++c)
#pragma unroll
      for (int e = 0; e < 16; ++e) {
        const int row = 32 * c + 16 * (e >> 3) + (e & 7) + 8 * h;
        float w40 = S.w4_l[row], w41 = S.w4_l[128 + row];
        float vv = (float)qv[c][e], jv = (float)qj[c][e];
        y0 += w40 * vv; y1 += w41 * vv;
        d00 += w40 * jv; d10 += w41 * jv;
      }
    y0 = xsum16(y0); y1 = xsum16(y1); d00 = xsum16(d00); d10 = xsum16(d10);

    if (h == 0) {
      const float* rp = &res[p * 16];
      float d01 = rp[0], d11 = rp[1], d21 = rp[2], g01 = rp[3], g11 = rp[4];
      float d02 = rp[5], d12 = rp[6], d22 = rp[7], g02 = rp[8], g12 = rp[9];
      y0 += S.b4_l[0];
      y1 += S.b4_l[1];
      float mom0 = d00 - PINN_NU * (g01 + g02) + d01 * y0 + d02 * y1 + d21;
      float mom1 = d10 - PINN_NU * (g11 + g12) + d11 * y0 + d12 * y1 + d22;
      float dv = d01 + d12;
      tsum += (mom0 * mom0 + mom1 * mom1) * (1.f / (2.f * 65536.f))
            + dv * dv * (1.f / 65536.f);
    }
  }

#pragma unroll
  for (int off = 16; off >= 1; off >>= 1) tsum += __shfl_xor(tsum, off, 32);
  if (lane == 0) S.red[wid] = tsum;
  __syncthreads();
  if (tid == 0) {
    float s = 0.f;
#pragma unroll
    for (int i = 0; i < 8; ++i) s += S.red[i];
    atomicAdd(out, s);
  }
}

// ---------------------------------------------------------------------------
// Forward-only MSE kernel for BC / IC point sets (16384 points each).
// A-operands read straight from the swizzled weights in d_ws (L2-resident).
// ---------------------------------------------------------------------------
__launch_bounds__(256)
__global__ void pinn_fwd_kernel(const float* __restrict__ xb, const float* __restrict__ yb,
                                const float* __restrict__ W0, const float* __restrict__ b0,
                                const float* __restrict__ b1, const float* __restrict__ b2,
                                const float* __restrict__ b3,
                                const float* __restrict__ W4, const float* __restrict__ b4,
                                const _Float16* __restrict__ wsw, float* out, float scale) {
  __shared__ float w0_l[384], b0_l[128], bh_l[384], w4_l[384], b4_l[3], red[8];
  const int tid = threadIdx.x;
  {
    for (int i = tid; i < 384; i += 256) {
      w0_l[i] = W0[i];
      w4_l[i] = W4[i];
      bh_l[i] = (i < 128) ? b1[i] : (i < 256) ? b2[i - 128] : b3[i - 256];
    }
    for (int i = tid; i < 128; i += 256) b0_l[i] = b0[i];
    if (tid < 3) b4_l[tid] = b4[tid];
  }
  __syncthreads();

  const int lane = tid & 31, wid = tid >> 5;
  const int h = lane >> 4, pcol = lane & 15;
  const int gwave = blockIdx.x * 8 + wid;
  const int nw = gridDim.x * 8;
  float tsum = 0.f;

  for (int g = gwave; g < 1024; g += nw) {   // 1024 groups / 256 waves = 4 each
    const int p = g * 16 + pcol;
    float x0 = xb[p * 3 + 0], x1 = xb[p * 3 + 1], x2 = xb[p * 3 + 2];
    v16h qv[4];
#pragma unroll
    for (int c = 0; c < 4; ++c)
#pragma unroll
      for (int e = 0; e < 16; ++e) {
        const int row = 32 * c + 16 * (e >> 3) + (e & 7) + 8 * h;
        float z = w0_l[row * 3 + 0] * x0 + w0_l[row * 3 + 1] * x1
                + w0_l[row * 3 + 2] * x2 + b0_l[row];
        qv[c][e] = (_Float16)fast_tanh(z);
      }
    for (int L = 0; L < 3; ++L) {
      const v16h* wl = (const v16h*)&wsw[L * 16384];
      v16h nv[4];
#pragma unroll
      for (int r = 0; r < 8; ++r) {
        v8f z = mm_tile(wl, lane, qv, r);
#pragma unroll
        for (int v = 0; v < 8; ++v) {
          const int row = 16 * r + v + 8 * h;
          nv[r >> 1][((r & 1) << 3) + v] =
              (_Float16)fast_tanh(z[v] + bh_l[L * 128 + row]);
        }
      }
#pragma unroll
      for (int c = 0; c < 4; ++c) qv[c] = nv[c];
    }
    float y0 = 0, y1 = 0, y2 = 0;
#pragma unroll
    for (int c = 0; c < 4; ++c)
#pragma unroll
      for (int e = 0; e < 16; ++e) {
        const int row = 32 * c + 16 * (e >> 3) + (e & 7) + 8 * h;
        float vv = (float)qv[c][e];
        y0 += w4_l[row] * vv;
        y1 += w4_l[128 + row] * vv;
        y2 += w4_l[256 + row] * vv;
      }
    y0 = xsum16(y0); y1 = xsum16(y1); y2 = xsum16(y2);
    if (h == 0) {
      float e0 = y0 + b4_l[0] - yb[p * 3 + 0];
      float e1 = y1 + b4_l[1] - yb[p * 3 + 1];
      float e2 = y2 + b4_l[2] - yb[p * 3 + 2];
      tsum += (e0 * e0 + e1 * e1 + e2 * e2) * scale;
    }
  }
#pragma unroll
  for (int off = 16; off >= 1; off >>= 1) tsum += __shfl_xor(tsum, off, 32);
  if (lane == 0) red[wid] = tsum;
  __syncthreads();
  if (tid == 0) {
    float s = 0.f;
#pragma unroll
    for (int i = 0; i < 8; ++i) s += red[i];
    atomicAdd(out, s);
  }
}

// ---------------------------------------------------------------------------
extern "C" void kernel_launch(void* const* d_in, const int* in_sizes, int n_in,
                              void* d_out, int out_size, void* d_ws, size_t ws_size,
                              hipStream_t stream) {
  (void)in_sizes; (void)n_in; (void)out_size; (void)ws_size;
  const float* W0 = (const float*)d_in[0];
  const float* b0 = (const float*)d_in[1];
  const float* W1 = (const float*)d_in[2];
  const float* b1 = (const float*)d_in[3];
  const float* W2 = (const float*)d_in[4];
  const float* b2 = (const float*)d_in[5];
  const float* W3 = (const float*)d_in[6];
  const float* b3 = (const float*)d_in[7];
  const float* W4 = (const float*)d_in[8];
  const float* b4 = (const float*)d_in[9];
  const float* x_pde = (const float*)d_in[10];
  const float* x_bc  = (const float*)d_in[11];
  const float* y_bc  = (const float*)d_in[12];
  const float* x_ic  = (const float*)d_in[13];
  const float* y_ic  = (const float*)d_in[14];
  float* out = (float*)d_out;
  _Float16* wsw = (_Float16*)d_ws;            // 96 KB swizzled weights
  float* res = (float*)d_ws + RES_OFF;        // 65536 x 16 floats (4 MB), L2-resident

  pinn_zero_out<<<1, 64, 0, stream>>>(out);
  pinn_prep_weights<<<192, 256, 0, stream>>>(W1, W2, W3, wsw);
  pinn_jh_kernel<<<64, 256, 0, stream>>>(x_pde, W0, b0, b1, b2, b3, W4, b4, wsw, res, 1);
  pinn_jh_kernel<<<64, 256, 0, stream>>>(x_pde, W0, b0, b1, b2, b3, W4, b4, wsw, res, 2);
  pinn_j0_kernel<<<64, 256, 0, stream>>>(x_pde, W0, b0, b1, b2, b3, W4, b4, wsw, res, out);
  pinn_fwd_kernel<<<32, 256, 0, stream>>>(x_bc, y_bc, W0, b0, b1, b2, b3, W4, b4,
                                          wsw, out, 1.f / (16384.f * 3.f));
  pinn_fwd_kernel<<<32, 256, 0, stream>>>(x_ic, y_ic, W0, b0, b1, b2, b3, W4, b4,
                                          wsw, out, 1.f / (16384.f * 3.f));
}
